// ProbTravelTime_spatial_hybrid_25134148616288
// MI455X (gfx1250) — matrix operand order repaired and checked by reading, not compiled, not run on hardware
//
#include <hip/hip_runtime.h>
#include <hip/hip_bf16.h>

// ---- shapes (fixed by the reference) ----
#define BB    256
#define SS    512
#define DR    256
#define DC    128
#define KIN   512      // DR + DC + 128
#define HID   512
#define NROWS (BB*SS)  // 131072
#define MT    64       // rows per block in main GEMM
#define NT    512      // threads per block in main GEMM (16 waves)
#define LDA   520      // bf16 elems per A row in LDS (1040B, 16B aligned)
#define LDB   40       // bf16 elems per B row in LDS (80B, 16B aligned)

typedef __attribute__((ext_vector_type(16))) __bf16 v16bf;
typedef __attribute__((ext_vector_type(8)))  float  v8f;

union ABFrag { v16bf v; uint4 u[2]; };

__device__ __forceinline__ unsigned short f2b(float f) {
  // fp32 -> bf16 round-to-nearest-even
  unsigned int u = __float_as_uint(f);
  unsigned int r = u + 0x7FFFu + ((u >> 16) & 1u);
  return (unsigned short)(r >> 16);
}

__device__ __forceinline__ float selu_f(float x) {
  const float a  = 1.6732632423543772f;
  const float sc = 1.0507009873554805f;
  return sc * (x > 0.f ? x : a * (__expf(x) - 1.f));
}

// ---- kernel 0: W1 (512x512 f32, [k][n]) -> W1T bf16 [n][k] ----
__global__ void __launch_bounds__(256) k_prep_w1t(const float* __restrict__ W1,
                                                  unsigned short* __restrict__ W1T) {
  int idx = blockIdx.x * 256 + threadIdx.x;   // idx = n*512 + k
  int n = idx >> 9;
  int k = idx & 511;
  W1T[idx] = f2b(W1[k * HID + n]);
}

// ---- kernel 1: ragged gather-mean embed + f2 selu MLP -> c_stack (B,128) ----
__global__ void __launch_bounds__(256) k_embed_f2(
    const float* __restrict__ cmap,   // (1,128,17,17) -> cmap[c*289 + h*17 + w]
    const int* __restrict__ lat, const int* __restrict__ lon,
    const float* __restrict__ W1, const float* __restrict__ bb1,
    const float* __restrict__ W2, const float* __restrict__ bb2,
    float* __restrict__ cstack) {
  __shared__ int   sidx[SS];
  __shared__ float semb[128];
  __shared__ float shdn[256];
  const int b = blockIdx.x, tid = threadIdx.x;

  for (int j = tid; j < SS; j += 256)
    sidx[j] = lat[b * SS + j] * 17 + lon[b * SS + j];
  __syncthreads();

  if (tid < 128) {
    const float* base = cmap + tid * 289;
    float s = 0.f;
    for (int j = 0; j < SS; ++j) s += base[sidx[j]];
    semb[tid] = s * (1.0f / 512.0f);
  }
  __syncthreads();

  {  // hidden layer: 256 units, selu
    float a = bb1[tid];
    for (int k = 0; k < 128; ++k) a = fmaf(semb[k], W1[k * 256 + tid], a);
    shdn[tid] = selu_f(a);
  }
  __syncthreads();

  if (tid < 128) {
    float a = bb2[tid];
    for (int j = 0; j < 256; ++j) a = fmaf(shdn[j], W2[j * 128 + tid], a);
    cstack[b * 128 + tid] = a;
  }
}

// ---- kernel 2: main fused layer ----
// x = [rho | c | c_stack[b]]  (row = b*512+s, K=512)
// h1 = selu(x @ W1 + b1); logm = h1.W21 + b21; logv = h1.W22 + b22
// writes lmw[row] = logm + log(w), lvw[row] = logv + 2*log(w)
__global__ void __launch_bounds__(NT) k_main(
    const float* __restrict__ rho, const float* __restrict__ cvec,
    const unsigned short* __restrict__ W1T,   // bf16 [n][k]
    const float* __restrict__ cstack,
    const float* __restrict__ b1,  const float* __restrict__ W21,
    const float* __restrict__ b21p, const float* __restrict__ W22,
    const float* __restrict__ b22p, const float* __restrict__ wgt,
    float* __restrict__ lmw, float* __restrict__ lvw) {
  extern __shared__ char smem[];
  unsigned short* sA  = (unsigned short*)smem;        // MT x LDA bf16
  unsigned short* sB0 = sA + MT * LDA;                // 512 x LDB bf16 (K-slice, buf 0)
  unsigned short* sB1 = sB0 + 512 * LDB;              // buf 1
  float* redm = (float*)(smem + (size_t)(MT * LDA + 2 * 512 * LDB) * 2);
  float* redv = redm + MT;

  const int tid  = threadIdx.x;
  const int wave = tid >> 5, lane = tid & 31;
  const int l16  = lane & 15, hh = lane >> 4;        // ISA lane-half split
  const int wm   = wave >> 2, wn = wave & 3;         // 4x4 wave grid (M x N)
  const int r0   = blockIdx.x * MT;                  // 64 rows, all same batch b
  const int b    = r0 >> 9;

  // Async L2->LDS staging of W1T[:, k0:k0+32): 512 rows x 64B, 4 b128 per thread,
  // 4 consecutive lanes cover one row's 64B chunk (coalesced).
  auto stageB = [&](unsigned short* buf, int k0) {
#pragma unroll
    for (int q8 = 0; q8 < 4; ++q8) {
      int n = (tid >> 2) + q8 * 128;
      int q = tid & 3;
      const unsigned short* gsrc = W1T + (size_t)n * KIN + k0 + q * 8;
      unsigned ldsa = (unsigned)(uintptr_t)(buf + n * LDB + q * 8);
      asm volatile("global_load_async_to_lds_b128 %0, %1, off"
                   :: "v"(ldsa), "v"(gsrc) : "memory");
    }
  };

  v8f acc[8];
#pragma unroll
  for (int t = 0; t < 8; ++t)
#pragma unroll
    for (int r = 0; r < 8; ++r) acc[t][r] = 0.f;

  // 8 WMMAs on one staged 32-wide K slice; acc updated in place.
  auto computeStep = [&](const unsigned short* bbuf, int k0) {
    // A fragment, 16x32 bf16 per ISA layout:
    //   elems 0..7 -> K = k0 + 8*half + j ; elems 8..15 -> K = k0 + 16 + 8*half + j
    const unsigned short* arow = sA + (wm * 16 + l16) * LDA + k0;
    ABFrag af;
    af.u[0] = *(const uint4*)(arow + 8 * hh);
    af.u[1] = *(const uint4*)(arow + 16 + 8 * hh);
#pragma unroll
    for (int t = 0; t < 8; ++t) {
      // B fragment 32x16: lane n = l16, elems 0..7 -> K = k0 + 8*half + j ; elems 8..15 -> K = k0 + 16 + 8*half + j
      const unsigned short* brow = bbuf + (wn * 128 + t * 16 + l16) * LDB + 8 * hh;
      ABFrag bf;
      bf.u[0] = *(const uint4*)(brow);
      bf.u[1] = *(const uint4*)(brow + 16);
      acc[t] = __builtin_amdgcn_wmma_f32_16x16x32_bf16(
          false, af.v, false, bf.v, (short)0, acc[t], false, false);
    }
  };

  // Kick off first B slice before doing the (VALU-heavy) A conversion.
  stageB(sB0, 0);

  // Stage A tile (x rows) as bf16 into LDS, vectorized float4 -> ushort4.
  // rho region: 64 rows x 256 cols
  for (int e = tid; e < MT * (DR / 4); e += NT) {
    int i = e >> 6, k4 = (e & 63) * 4;
    float4 v = *(const float4*)(rho + (size_t)(r0 + i) * DR + k4);
    ushort4 o; o.x = f2b(v.x); o.y = f2b(v.y); o.z = f2b(v.z); o.w = f2b(v.w);
    *(ushort4*)(sA + i * LDA + k4) = o;
  }
  // c region: cols 256..383 (same for all rows)
  for (int e = tid; e < MT * (DC / 4); e += NT) {
    int i = e >> 5, k4 = (e & 31) * 4;
    float4 v = *(const float4*)(cvec + k4);
    ushort4 o; o.x = f2b(v.x); o.y = f2b(v.y); o.z = f2b(v.z); o.w = f2b(v.w);
    *(ushort4*)(sA + i * LDA + DR + k4) = o;
  }
  // c_stack region: cols 384..511 (same for all rows of this batch)
  for (int e = tid; e < MT * (128 / 4); e += NT) {
    int i = e >> 5, k4 = (e & 31) * 4;
    float4 v = *(const float4*)(cstack + b * 128 + k4);
    ushort4 o; o.x = f2b(v.x); o.y = f2b(v.y); o.z = f2b(v.z); o.w = f2b(v.w);
    *(ushort4*)(sA + i * LDA + DR + DC + k4) = o;
  }
  if (tid < 2 * MT) redm[tid] = 0.f;   // zero redm[64] + redv[64]

  asm volatile("s_wait_asynccnt 0x0" ::: "memory");
  __syncthreads();

  // Manually 2x-unrolled K loop: buffer pointers are compile-time constants in
  // each half, so accumulators keep a single register home (no v_mov rotation).
#pragma unroll 1
  for (int kk2 = 0; kk2 < KIN / 64; ++kk2) {
    const int k0 = kk2 * 64;

    stageB(sB1, k0 + 32);                 // prefetch odd slice
    computeStep(sB0, k0);                 // compute even slice
    asm volatile("s_wait_asynccnt 0x0" ::: "memory");
    __syncthreads();

    if (kk2 + 1 < KIN / 64)
      stageB(sB0, k0 + 64);               // prefetch next even slice
    computeStep(sB1, k0 + 32);            // compute odd slice
    asm volatile("s_wait_asynccnt 0x0" ::: "memory");
    __syncthreads();
  }

  // Epilogue: selu + dot with W21/W22, reduce across lanes/waves via LDS atomics.
  // C/D layout: VGPR r -> m = 8*half + r ; n = l16  (per wave tile)
  const float b21 = b21p[0], b22 = b22p[0];
  float sm[8], sv[8];
#pragma unroll
  for (int r = 0; r < 8; ++r) { sm[r] = 0.f; sv[r] = 0.f; }
#pragma unroll
  for (int t = 0; t < 8; ++t) {
    int n = wn * 128 + t * 16 + l16;
    float w1b = b1[n], w21v = W21[n], w22v = W22[n];
#pragma unroll
    for (int r = 0; r < 8; ++r) {
      float h = selu_f(acc[t][r] + w1b);
      sm[r] = fmaf(h, w21v, sm[r]);
      sv[r] = fmaf(h, w22v, sv[r]);
    }
  }
#pragma unroll
  for (int r = 0; r < 8; ++r) {
    int mi = wm * 16 + hh * 8 + r;
    atomicAdd(&redm[mi], sm[r]);   // ds_add_f32
    atomicAdd(&redv[mi], sv[r]);
  }
  __syncthreads();

  if (tid < MT) {
    int row = r0 + tid;
    float lw = __logf(wgt[row]);
    lmw[row] = redm[tid] + b21 + lw;
    lvw[row] = redv[tid] + b22 + 2.f * lw;
  }
}

// ---- kernel 3: per-batch logsumexp over S, final outputs ----
__global__ void __launch_bounds__(256) k_reduce(const float* __restrict__ lmw,
                                                const float* __restrict__ lvw,
                                                const float* __restrict__ l,
                                                float* __restrict__ out) {
  __shared__ float sm[256], sv[256];
  const int b = blockIdx.x, tid = threadIdx.x;
  const float* pm = lmw + (size_t)b * SS;
  const float* pv = lvw + (size_t)b * SS;

  sm[tid] = fmaxf(pm[tid], pm[tid + 256]);
  sv[tid] = fmaxf(pv[tid], pv[tid + 256]);
  __syncthreads();
  for (int st = 128; st > 0; st >>= 1) {
    if (tid < st) {
      sm[tid] = fmaxf(sm[tid], sm[tid + st]);
      sv[tid] = fmaxf(sv[tid], sv[tid + st]);
    }
    __syncthreads();
  }
  float mm = sm[0], mv = sv[0];
  __syncthreads();

  sm[tid] = __expf(pm[tid] - mm) + __expf(pm[tid + 256] - mm);
  sv[tid] = __expf(pv[tid] - mv) + __expf(pv[tid + 256] - mv);
  __syncthreads();
  for (int st = 128; st > 0; st >>= 1) {
    if (tid < st) { sm[tid] += sm[tid + st]; sv[tid] += sv[tid + st]; }
    __syncthreads();
  }
  if (tid == 0) {
    float lmagg = mm + __logf(sm[0]);
    float lvagg = mv + __logf(sv[0]);
    float ll = __logf(l[b]);
    out[b]       = ll - lmagg;                       // log_mu
    out[256 + b] = ll - 3.f * lmagg - lvagg;         // log_lambda
  }
}

extern "C" void kernel_launch(void* const* d_in, const int* in_sizes, int n_in,
                              void* d_out, int out_size, void* d_ws, size_t ws_size,
                              hipStream_t stream) {
  const float* rho   = (const float*)d_in[0];
  const float* cvec  = (const float*)d_in[1];
  const float* cmap  = (const float*)d_in[2];
  const float* w     = (const float*)d_in[3];
  const float* l     = (const float*)d_in[4];
  // d_in[5] roads: unused by the reference math
  const int*   lat   = (const int*)d_in[6];
  const int*   lon   = (const int*)d_in[7];
  const float* f2W1  = (const float*)d_in[8];
  const float* f2b1  = (const float*)d_in[9];
  const float* f2W2  = (const float*)d_in[10];
  const float* f2b2  = (const float*)d_in[11];
  const float* fW1   = (const float*)d_in[12];
  const float* fb1   = (const float*)d_in[13];
  const float* fW21  = (const float*)d_in[14];
  const float* fb21  = (const float*)d_in[15];
  const float* fW22  = (const float*)d_in[16];
  const float* fb22  = (const float*)d_in[17];
  float* out = (float*)d_out;

  // workspace layout (~1.7 MB total)
  unsigned short* W1T = (unsigned short*)d_ws;                         // 512 KB
  float* cstack = (float*)((char*)d_ws + (size_t)HID * KIN * 2);       // 128 KB
  float* lmw = (float*)((char*)d_ws + (size_t)HID * KIN * 2 + (size_t)BB * 128 * 4);
  float* lvw = lmw + NROWS;                                            // 2 x 512 KB

  k_prep_w1t<<<(HID * KIN) / 256, 256, 0, stream>>>(fW1, W1T);
  k_embed_f2<<<BB, 256, 0, stream>>>(cmap, lat, lon, f2W1, f2b1, f2W2, f2b2, cstack);

  size_t shmem = (size_t)(MT * LDA + 2 * 512 * LDB) * 2 + 2 * MT * sizeof(float);
  k_main<<<NROWS / MT, NT, shmem, stream>>>(rho, cvec, W1T, cstack,
                                            fb1, fW21, fb21, fW22, fb22, w,
                                            lmw, lvw);
  k_reduce<<<BB, 256, 0, stream>>>(lmw, lvw, l, out);
}
